// ATDTransformerLayer_81398220194003
// MI455X (gfx1250) — matrix operand order repaired and checked by reading, not compiled, not run on hardware
//
#include <hip/hip_runtime.h>
#include <hip/hip_bf16.h>
#include <math.h>

// ---------------------------------------------------------------------------
// ATD transformer layer for gfx1250 (MI455X).  All matrix math through
// v_wmma_f32_16x16x32_f16 (wave32, 16x16 tiles, f32 accumulate).
// GEMM operands staged through LDS with 128-bit transfers (async-to-LDS via
// the gfx1250 ASYNCcnt path), B pre-transposed in LDS so every WMMA fragment
// read is a contiguous 32-byte LDS access.
// ---------------------------------------------------------------------------

typedef _Float16 half_t;
typedef __attribute__((ext_vector_type(16))) _Float16 v16h;
typedef __attribute__((ext_vector_type(8)))  _Float16 v8h;
typedef __attribute__((ext_vector_type(8)))  float    v8f;

static constexpr int kB     = 2;
static constexpr int kH     = 192;
static constexpr int kW     = 192;
static constexpr int kN     = 36864;       // tokens per batch
static constexpr int kBN    = 73728;       // B*N
static constexpr int kC     = 192;
static constexpr int kNH    = 6;
static constexpr int kHD    = 32;
static constexpr int kNT    = 256;         // window / group size
static constexpr int kNWIN  = 144;         // windows (== groups) per batch
static constexpr int kM     = 64;          // token dictionary size
static constexpr int kRC    = 10;
static constexpr int kHID   = 384;
static constexpr int kSHIFT = 8;

__device__ __forceinline__ float gelu_f(float x) {
  return 0.5f * x * (1.0f + erff(x * 0.70710678118654752f));
}

// ---------------------------------------------------------------------------
// CDNA5 async global->LDS copy (ASYNCcnt-tracked), with portable fallback.
// Builtin signature on this toolchain: (int4 AS1*, int4 AS3*, imm, imm).
// ---------------------------------------------------------------------------
#if __has_builtin(__builtin_amdgcn_global_load_async_to_lds_b128) && \
    __has_builtin(__builtin_amdgcn_s_wait_asynccnt)
#define ATD_ASYNC_LDS 1
#else
#define ATD_ASYNC_LDS 0
#endif

#if ATD_ASYNC_LDS
typedef int v4i_vs __attribute__((vector_size(16)));
typedef __attribute__((address_space(1))) v4i_vs* gptr_v4i;
typedef __attribute__((address_space(3))) v4i_vs* lptr_v4i;
#endif

__device__ __forceinline__ void copy16B_g2l(const half_t* g, half_t* l) {
#if ATD_ASYNC_LDS
  // global: flat and AS1 addresses coincide -> integer round-trip is exact.
  // lds: must go through an addrspacecast (aperture-relative offsets).
  __builtin_amdgcn_global_load_async_to_lds_b128(
      (gptr_v4i)(size_t)g, (lptr_v4i)(v4i_vs*)l, 0, 0);
#else
  *(v8h*)l = *(const v8h*)g;
#endif
}

__device__ __forceinline__ void async_fence() {
#if ATD_ASYNC_LDS
  __builtin_amdgcn_s_wait_asynccnt(0);
#endif
}

// ---------------------------------------------------------------------------
// WMMA fragment loaders (contiguous k = 16*(lane/16) + e in both A and B, wave32)
// ---------------------------------------------------------------------------

// A fragment from a row-major 16 x lda f16 tile in LDS.
__device__ __forceinline__ v16h ldsA_frag(const half_t* __restrict__ A, int lda,
                                          int lane) {
  int r = lane & 15;
  int kb = (lane & 16) ? 16 : 0;
  const half_t* p = A + r * lda + kb;
  v16h a;
#pragma unroll
  for (int i = 0; i < 8; ++i) { a[i] = p[i]; a[i + 8] = p[i + 8]; }
  return a;
}

// B fragment representing K^T : element (d, col) = Ksrc[col][d], Ksrc 256x32 LDS
__device__ __forceinline__ v16h ldsKT_frag(const half_t* __restrict__ K, int kt,
                                           int lane) {
  const half_t* p = K + (size_t)(kt * 16 + (lane & 15)) * 32 + ((lane & 16) ? 16 : 0);
  v16h a;
#pragma unroll
  for (int i = 0; i < 16; ++i) a[i] = p[i];
  return a;
}

// B fragment from a TRANSPOSED operand: BT[col][k] with leading dim ldt.
// Element e of the fragment is B[k0+kb+e][col0+lane15] = BT[col][k0+kb+e],
// i.e. 16 contiguous halves -> one 32-byte LDS read per lane.
__device__ __forceinline__ v16h ldsBT_frag(const half_t* __restrict__ BT, int ldt,
                                           int k0, int col0, int lane) {
  const half_t* p = BT + (size_t)(col0 + (lane & 15)) * ldt + k0 +
                    ((lane & 16) ? 16 : 0);
  v16h a;
#pragma unroll
  for (int e = 0; e < 16; ++e) a[e] = p[e];
  return a;
}

__device__ __forceinline__ v8f zero8() {
  v8f z = {0.f, 0.f, 0.f, 0.f, 0.f, 0.f, 0.f, 0.f};
  return z;
}

// ---------------------------------------------------------------------------
// Generic f16 GEMM:  C[M x N] = act(A[M x K] @ B[K x N] + bias)  (+= if accum)
// block = 128 threads (4 waves); block tile 64x64; wave tile 16x64.
// REQUIRES: Mr % 64 == 0, Nc % 64 == 0, K % 32 == 0 (all call sites comply).
// ---------------------------------------------------------------------------
__global__ __launch_bounds__(128) void gemm_f16_kernel(
    const half_t* __restrict__ A, const half_t* __restrict__ Bm,
    const float* __restrict__ bias, float* __restrict__ Cm,
    int Mr, int Nc, int K, int act, int accum) {
  (void)Mr;
  int lane = threadIdx.x & 31;
  int wave = threadIdx.x >> 5;
  int tid = threadIdx.x;
  int rowblk = blockIdx.y * 64;
  int col0 = blockIdx.x * 64;

  __shared__ alignas(16) half_t sA[64 * 32];   // [r][k]   row-major slab
  __shared__ alignas(16) half_t sBT[64 * 32];  // [c][k]   transposed slab

  v8f acc[4];
#pragma unroll
  for (int t = 0; t < 4; ++t) acc[t] = zero8();

  int bk = tid >> 2;          // B row within slab (0..31)
  int bc = (tid & 3) * 16;    // B col segment start (0,16,32,48)

  for (int k0 = 0; k0 < K; k0 += 32) {
    // ---- stage A slab (64x32) with 16-byte async copies ----
#pragma unroll
    for (int it = 0; it < 2; ++it) {
      int ch = tid + it * 128;          // 256 chunks of 8 halves
      int r = ch >> 2, off = (ch & 3) * 8;
      copy16B_g2l(A + (size_t)(rowblk + r) * K + k0 + off, &sA[r * 32 + off]);
    }
    // ---- stage B slab transposed: wide global reads, 16-bit LDS scatter ----
    {
      const half_t* gp = Bm + (size_t)(k0 + bk) * Nc + col0 + bc;
      if (k0 + 64 < K) __builtin_prefetch(gp + (size_t)32 * Nc, 0, 1);
      v8h b0 = *(const v8h*)gp;
      v8h b1 = *(const v8h*)(gp + 8);
#pragma unroll
      for (int j = 0; j < 8; ++j) {
        sBT[(bc + j) * 32 + bk] = b0[j];
        sBT[(bc + 8 + j) * 32 + bk] = b1[j];
      }
    }
    async_fence();
    __syncthreads();

    v16h a = ldsA_frag(sA + wave * 16 * 32, 32, lane);
#pragma unroll
    for (int t = 0; t < 4; ++t) {
      v16h b = ldsBT_frag(sBT, 32, 0, t * 16, lane);
      acc[t] = __builtin_amdgcn_wmma_f32_16x16x32_f16(false, a, false, b,
                                                      (short)0, acc[t], false, false);
    }
    __syncthreads();
  }

  int colb = lane & 15;
  int rb = rowblk + wave * 16 + ((lane & 16) ? 8 : 0);
#pragma unroll
  for (int t = 0; t < 4; ++t) {
    int c0 = col0 + t * 16 + colb;
#pragma unroll
    for (int r = 0; r < 8; ++r) {
      float vv = acc[t][r];
      if (bias) vv += bias[c0];
      if (act == 1) vv = gelu_f(vv);
      size_t idx = (size_t)(rb + r) * Nc + c0;
      if (accum) Cm[idx] += vv; else Cm[idx] = vv;
    }
  }
}

// ---------------------------------------------------------------------------
// Fused 256-token attention (shifted-window MSA and AC-MSA).
// One block = one (unit, head).  block = 128 threads (4 waves), each wave owns
// 4 strips of 16 query rows; flash-style 2-pass softmax, scores never hit HBM.
// V is stored TRANSPOSED in LDS so P@V B-fragments are contiguous reads.
// ---------------------------------------------------------------------------
template <int WIN>
__global__ __launch_bounds__(128) void attn256_kernel(
    const float* __restrict__ qkv, const int* __restrict__ sort_idx,
    const int* __restrict__ rpi, const float* __restrict__ amask,
    const float* __restrict__ rpb_table, const float* __restrict__ logit_scale,
    half_t* __restrict__ out_h) {
  int head = blockIdx.x % kNH;
  int unit = blockIdx.x / kNH;          // window id (b*144+wl) or group id
  int lane = threadIdx.x & 31;
  int wave = threadIdx.x >> 5;
  int b = unit / kNWIN;
  int wl = unit % kNWIN;

  __shared__ alignas(16) half_t sq[kNT * kHD];    // [t][d]
  __shared__ alignas(16) half_t sk[kNT * kHD];    // [t][d]
  __shared__ alignas(16) half_t svT[kHD * kNT];   // [d][t]  transposed
  __shared__ alignas(16) half_t pb[4][16 * 32];
  __shared__ int srow[kNT];
  __shared__ int sout[kNT];

  for (int t = threadIdx.x; t < kNT; t += blockDim.x) {
    int row, orow;
    if (WIN) {
      int hh = (wl / 12) * 16 + (t >> 4);
      int ww = (wl % 12) * 16 + (t & 15);
      int oh = (hh + kSHIFT) % kH;      // undo roll(-SHIFT) by reading +SHIFT
      int ow = (ww + kSHIFT) % kW;
      row = b * kN + oh * kW + ow;
      orow = unit * kNT + t;
    } else {
      int orig = sort_idx[b * kN + wl * kNT + t];
      row = b * kN + orig;
      orow = b * kN + orig;              // write back unsorted directly
    }
    srow[t] = row;
    sout[t] = orow;
  }
  __syncthreads();

  float qscale = WIN ? 0.17677669529663687f : 1.0f;  // hd^-0.5
  float ls = 1.0f;
  if (!WIN) ls = __expf(fminf(logit_scale[0], 4.605170185988091f));

  for (int i = threadIdx.x; i < kNT * kHD; i += blockDim.x) {
    int t = i >> 5, d = i & 31;
    size_t base = (size_t)srow[t] * (3 * kC) + head * kHD + d;
    sq[i] = (half_t)(qkv[base] * qscale);
    sk[i] = (half_t)(qkv[base + kC]);
    svT[d * kNT + t] = (half_t)(qkv[base + 2 * kC]);
  }
  __syncthreads();

  for (int s = 0; s < 4; ++s) {
    int strip = wave * 4 + s;
    int row0 = strip * 16;
    v16h qa = ldsA_frag(sq + (size_t)row0 * 32, 32, lane);

    // ---- pass 1: row max (bias included) ----
    float mx[8];
#pragma unroll
    for (int r = 0; r < 8; ++r) mx[r] = -3.0e38f;
    for (int kt = 0; kt < 16; ++kt) {
      v16h kb = ldsKT_frag(sk, kt, lane);
      v8f sc = zero8();
      sc = __builtin_amdgcn_wmma_f32_16x16x32_f16(false, qa, false, kb,
                                                  (short)0, sc, false, false);
#pragma unroll
      for (int r = 0; r < 8; ++r) {
        float v = sc[r] * ls;
        if (WIN) {
          int tq = row0 + r + ((lane & 16) ? 8 : 0);
          int tu = kt * 16 + (lane & 15);
          v += rpb_table[rpi[tq * kNT + tu] * kNH + head] +
               amask[((size_t)wl * kNT + tq) * kNT + tu];
        }
        mx[r] = fmaxf(mx[r], v);
      }
    }
#pragma unroll
    for (int r = 0; r < 8; ++r)
      for (int off = 1; off < 16; off <<= 1)
        mx[r] = fmaxf(mx[r], __shfl_xor(mx[r], off, 32));

    // ---- pass 2: recompute, exp, P@V ----
    v8f o0 = zero8(), o1 = zero8();
    float lsum[8];
#pragma unroll
    for (int r = 0; r < 8; ++r) lsum[r] = 0.f;

    for (int k2 = 0; k2 < 8; ++k2) {
#pragma unroll
      for (int sub = 0; sub < 2; ++sub) {
        int kt = k2 * 2 + sub;
        v16h kb = ldsKT_frag(sk, kt, lane);
        v8f sc = zero8();
        sc = __builtin_amdgcn_wmma_f32_16x16x32_f16(false, qa, false, kb,
                                                    (short)0, sc, false, false);
#pragma unroll
        for (int r = 0; r < 8; ++r) {
          float v = sc[r] * ls;
          if (WIN) {
            int tq = row0 + r + ((lane & 16) ? 8 : 0);
            int tu = kt * 16 + (lane & 15);
            v += rpb_table[rpi[tq * kNT + tu] * kNH + head] +
                 amask[((size_t)wl * kNT + tq) * kNT + tu];
          }
          float p = __expf(v - mx[r]);
          lsum[r] += p;
          int prow = r + ((lane & 16) ? 8 : 0);
          int pcol = sub * 16 + (lane & 15);
          pb[wave][prow * 32 + pcol] = (half_t)p;
        }
      }
      __syncthreads();   // C-layout -> A-layout bounce visible across lanes
      v16h pa = ldsA_frag(pb[wave], 32, lane);
      v16h vb0 = ldsBT_frag(svT, kNT, k2 * 32, 0, lane);
      v16h vb1 = ldsBT_frag(svT, kNT, k2 * 32, 16, lane);
      o0 = __builtin_amdgcn_wmma_f32_16x16x32_f16(false, pa, false, vb0,
                                                  (short)0, o0, false, false);
      o1 = __builtin_amdgcn_wmma_f32_16x16x32_f16(false, pa, false, vb1,
                                                  (short)0, o1, false, false);
      __syncthreads();   // reads done before next iteration overwrites pb
    }

#pragma unroll
    for (int r = 0; r < 8; ++r)
      for (int off = 1; off < 16; off <<= 1)
        lsum[r] += __shfl_xor(lsum[r], off, 32);

    int col = lane & 15;
#pragma unroll
    for (int r = 0; r < 8; ++r) {
      int t = row0 + r + ((lane & 16) ? 8 : 0);
      float inv = 1.0f / lsum[r];
      size_t obase = (size_t)sout[t] * kC + head * kHD;
      out_h[obase + col] = (half_t)(o0[r] * inv);
      out_h[obase + 16 + col] = (half_t)(o1[r] * inv);
    }
  }
}

// ---------------------------------------------------------------------------
// LayerNorm: one block (256 thr) per row of 192
// ---------------------------------------------------------------------------
__global__ __launch_bounds__(256) void ln_kernel(
    const float* __restrict__ x, const float* __restrict__ g,
    const float* __restrict__ bta, float* __restrict__ outf,
    half_t* __restrict__ outh) {
  int row = blockIdx.x, tid = threadIdx.x;
  __shared__ float red[256];
  float v = (tid < kC) ? x[(size_t)row * kC + tid] : 0.0f;
  red[tid] = v;
  __syncthreads();
  for (int off = 128; off; off >>= 1) {
    if (tid < off) red[tid] += red[tid + off];
    __syncthreads();
  }
  float mu = red[0] / (float)kC;
  __syncthreads();
  float d = (tid < kC) ? v - mu : 0.0f;
  red[tid] = d * d;
  __syncthreads();
  for (int off = 128; off; off >>= 1) {
    if (tid < off) red[tid] += red[tid + off];
    __syncthreads();
  }
  float var = red[0] / (float)kC;
  float rs = rsqrtf(var + 1e-5f);
  if (tid < kC) {
    float o = d * rs * g[tid] + bta[tid];
    if (outf) outf[(size_t)row * kC + tid] = o;
    if (outh) outh[(size_t)row * kC + tid] = (half_t)o;
  }
}

__global__ void f2h_kernel(const float* __restrict__ in, half_t* __restrict__ out,
                           int n) {
  int i = blockIdx.x * blockDim.x + threadIdx.x;
  if (i < n) out[i] = (half_t)in[i];
}

__global__ void fill0_kernel(int* p, int n) {
  int i = blockIdx.x * blockDim.x + threadIdx.x;
  if (i < n) p[i] = 0;
}

// q/k projection to RC=10 + L2 normalize (too narrow for WMMA; VALU is fine)
__global__ void proj_rc_norm_kernel(const float* __restrict__ A,
                                    const float* __restrict__ W,
                                    const float* __restrict__ bias,
                                    float* __restrict__ out, int rows) {
  int r = blockIdx.x * blockDim.x + threadIdx.x;
  if (r >= rows) return;
  float acc[kRC];
#pragma unroll
  for (int j = 0; j < kRC; ++j) acc[j] = 0.f;
  const float* a = A + (size_t)r * kC;
  for (int k = 0; k < kC; ++k) {
    float av = a[k];
#pragma unroll
    for (int j = 0; j < kRC; ++j) acc[j] += av * W[k * kRC + j];
  }
  float nrm = 0.f;
#pragma unroll
  for (int j = 0; j < kRC; ++j) { acc[j] += bias[j]; nrm += acc[j] * acc[j]; }
  nrm = fmaxf(sqrtf(nrm), 1e-12f);
  float inv = 1.0f / nrm;
#pragma unroll
  for (int j = 0; j < kRC; ++j) out[(size_t)r * kRC + j] = acc[j] * inv;
}

// ATD cross-attention scores + softmax over m=64 + argmax (tk_id)
__global__ __launch_bounds__(256) void atd_attn_kernel(
    const float* __restrict__ qn, const float* __restrict__ kn,
    const float* __restrict__ atd_scale, float* __restrict__ sim,
    half_t* __restrict__ sim_h, int* __restrict__ tk_id) {
  int idx = blockIdx.x * blockDim.x + threadIdx.x;
  int b = idx / kN;
  __shared__ float skn[kM * kRC];
  __shared__ float ssc[kM];
  for (int i = threadIdx.x; i < kM * kRC; i += blockDim.x)
    skn[i] = kn[(size_t)b * kM * kRC + i];
  for (int i = threadIdx.x; i < kM; i += blockDim.x) {
    float s = fminf(fmaxf(atd_scale[i], 0.f), 1.f);
    ssc[i] = 1.0f + s * 4.1588830833596715f;  // log(64)
  }
  __syncthreads();
  if (idx >= kBN) return;
  float qv[kRC];
#pragma unroll
  for (int j = 0; j < kRC; ++j) qv[j] = qn[(size_t)idx * kRC + j];
  float a[kM];
  float mx = -3e38f;
  int mi = 0;
  for (int m = 0; m < kM; ++m) {
    float d = 0.f;
#pragma unroll
    for (int j = 0; j < kRC; ++j) d += qv[j] * skn[m * kRC + j];
    d *= ssc[m];
    a[m] = d;
    if (d > mx) { mx = d; mi = m; }
  }
  float sum = 0.f;
  for (int m = 0; m < kM; ++m) { a[m] = __expf(a[m] - mx); sum += a[m]; }
  float inv = 1.0f / sum;
  for (int m = 0; m < kM; ++m) {
    float v = a[m] * inv;
    sim[(size_t)idx * kM + m] = v;
    sim_h[(size_t)idx * kM + m] = (half_t)v;
  }
  tk_id[idx] = mi;
}

__global__ void hist_kernel(const int* __restrict__ tk, int* __restrict__ bins) {
  int idx = blockIdx.x * blockDim.x + threadIdx.x;
  if (idx >= kBN) return;
  int b = idx / kN;
  atomicAdd(&bins[b * kM + tk[idx]], 1);
}

__global__ void scan_kernel(const int* __restrict__ bins, int* __restrict__ base) {
  int b = threadIdx.x;
  if (b >= kB) return;
  int acc = 0;
  for (int m = 0; m < kM; ++m) { base[b * kM + m] = acc; acc += bins[b * kM + m]; }
}

// Stable counting-sort scatter (matches jnp.argsort stability). One block per
// (batch,key): scan tokens in order, block-wide prefix over match flags.
__global__ __launch_bounds__(256) void stable_scatter_kernel(
    const int* __restrict__ tk, const int* __restrict__ base,
    int* __restrict__ sort_idx) {
  int b = blockIdx.x / kM, key = blockIdx.x % kM;
  __shared__ int flags[256];
  __shared__ int run;
  if (threadIdx.x == 0) run = base[b * kM + key];
  __syncthreads();
  for (int start = 0; start < kN; start += 256) {
    int n = start + threadIdx.x;
    int f = (tk[b * kN + n] == key) ? 1 : 0;
    flags[threadIdx.x] = f;
    __syncthreads();
    for (int off = 1; off < 256; off <<= 1) {
      int v = (threadIdx.x >= off) ? flags[threadIdx.x - off] : 0;
      __syncthreads();
      flags[threadIdx.x] += v;
      __syncthreads();
    }
    int base_run = run;
    if (f) sort_idx[(size_t)b * kN + base_run + flags[threadIdx.x] - 1] = n;
    __syncthreads();
    if (threadIdx.x == 0) run = base_run + flags[255];
    __syncthreads();
  }
}

// x = shortcut + x_win(window-reverse+roll via index math) + x_atd + x_aca
__global__ void residual_sum_kernel(const float* __restrict__ x0,
                                    const float* __restrict__ owp,
                                    const float* __restrict__ x_atd,
                                    const float* __restrict__ x_aca,
                                    float* __restrict__ out) {
  size_t idx = (size_t)blockIdx.x * blockDim.x + threadIdx.x;
  if (idx >= (size_t)kBN * kC) return;
  int c = idx % kC;
  int pos = (idx / kC) % kN;
  int b = idx / ((size_t)kN * kC);
  int h = pos / kW, w = pos % kW;
  int hh = (h - kSHIFT + kH) % kH, ww = (w - kSHIFT + kW) % kW;
  int win = b * kNWIN + (hh / 16) * 12 + (ww / 16);
  int t = (hh % 16) * 16 + (ww % 16);
  float xw = owp[((size_t)win * kNT + t) * kC + c];
  out[idx] = x0[idx] + xw + x_atd[idx] + x_aca[idx];
}

// depthwise 5x5 conv over (H,W) per channel, gelu, residual into y3 (+f16 copy)
__global__ void dwconv_kernel(const float* __restrict__ y2,
                              const float* __restrict__ wgt,
                              const float* __restrict__ bias,
                              float* __restrict__ y3, half_t* __restrict__ y3h) {
  size_t idx = (size_t)blockIdx.x * blockDim.x + threadIdx.x;
  if (idx >= (size_t)kBN * kHID) return;
  int ch = idx % kHID;
  int pos = (idx / kHID) % kN;
  int b = idx / ((size_t)kN * kHID);
  int h = pos / kW, w = pos % kW;
  float acc = 0.f;
#pragma unroll
  for (int ky = 0; ky < 5; ++ky) {
    int ih = h + ky - 2;
    if (ih < 0 || ih >= kH) continue;
#pragma unroll
    for (int kx = 0; kx < 5; ++kx) {
      int iw = w + kx - 2;
      if (iw < 0 || iw >= kW) continue;
      acc += y2[((size_t)b * kN + ih * kW + iw) * kHID + ch] *
             wgt[ch * 25 + ky * 5 + kx];
    }
  }
  float o = y2[idx] + gelu_f(acc + bias[ch]);
  y3[idx] = o;
  y3h[idx] = (half_t)o;
}

// softmax over n (36864) of sim^T columns -> mask_soft rows (f16 A-matrix)
__global__ __launch_bounds__(256) void col_softmax_kernel(
    const float* __restrict__ sim, half_t* __restrict__ ms) {
  int b = blockIdx.x >> 6, m = blockIdx.x & 63;
  const float* col = sim + (size_t)b * kN * kM + m;
  __shared__ float red[256];
  int tid = threadIdx.x;
  float mx = -3e38f;
  for (int n = tid; n < kN; n += 256) mx = fmaxf(mx, col[(size_t)n * kM]);
  red[tid] = mx;
  __syncthreads();
  for (int off = 128; off; off >>= 1) {
    if (tid < off) red[tid] = fmaxf(red[tid], red[tid + off]);
    __syncthreads();
  }
  mx = red[0];
  __syncthreads();
  float sm = 0.f;
  for (int n = tid; n < kN; n += 256) sm += __expf(col[(size_t)n * kM] - mx);
  red[tid] = sm;
  __syncthreads();
  for (int off = 128; off; off >>= 1) {
    if (tid < off) red[tid] += red[tid + off];
    __syncthreads();
  }
  sm = red[0];
  float inv = 1.0f / sm;
  half_t* o = ms + ((size_t)b * kM + m) * kN;
  for (int n = tid; n < kN; n += 256)
    o[n] = (half_t)(__expf(col[(size_t)n * kM] - mx) * inv);
}

__global__ void td_new_kernel(const float* __restrict__ td,
                              const float* __restrict__ tdac,
                              const float* __restrict__ sigma,
                              float* __restrict__ out) {
  int idx = blockIdx.x * blockDim.x + threadIdx.x;
  if (idx >= kB * kM * kC) return;
  int m = (idx / kC) % kM;
  float s = 1.0f / (1.0f + __expf(-sigma[m]));
  out[idx] = s * td[idx] + (1.0f - s) * tdac[idx];
}

// ---------------------------------------------------------------------------
// Host orchestration
// ---------------------------------------------------------------------------
static inline char* bump(char*& p, size_t bytes) {
  char* r = p;
  p += (bytes + 255) & ~(size_t)255;
  return r;
}
static inline dim3 g1(size_t total, int blk) { return dim3((unsigned)((total + blk - 1) / blk)); }

extern "C" void kernel_launch(void* const* d_in, const int* in_sizes, int n_in,
                              void* d_out, int out_size, void* d_ws, size_t ws_size,
                              hipStream_t stream) {
  (void)in_sizes; (void)n_in; (void)out_size; (void)ws_size;
  const float* x        = (const float*)d_in[0];
  const float* td       = (const float*)d_in[1];
  const int*   rpi_sa   = (const int*)d_in[2];
  const float* amask    = (const float*)d_in[3];
  const float* n1g = (const float*)d_in[6],  *n1b = (const float*)d_in[7];
  const float* n2g = (const float*)d_in[8],  *n2b = (const float*)d_in[9];
  const float* n3g = (const float*)d_in[10], *n3b = (const float*)d_in[11];
  const float* wqkv_w = (const float*)d_in[12], *wqkv_b = (const float*)d_in[13];
  const float* wq_w = (const float*)d_in[14], *wq_b = (const float*)d_in[15];
  const float* wk_w = (const float*)d_in[16], *wk_b = (const float*)d_in[17];
  const float* wv_w = (const float*)d_in[18], *wv_b = (const float*)d_in[19];
  const float* atd_scale = (const float*)d_in[20];
  const float* paca_w = (const float*)d_in[21], *paca_b = (const float*)d_in[22];
  const float* logit_scale = (const float*)d_in[23];
  const float* rpb_table = (const float*)d_in[24];
  const float* pwin_w = (const float*)d_in[25], *pwin_b = (const float*)d_in[26];
  const float* fc1_w = (const float*)d_in[27], *fc1_b = (const float*)d_in[28];
  const float* dw_w = (const float*)d_in[29], *dw_b = (const float*)d_in[30];
  const float* fc2_w = (const float*)d_in[31], *fc2_b = (const float*)d_in[32];
  const float* sigma = (const float*)d_in[33];

  float* out_x = (float*)d_out;                  // (B, N, C)
  float* out_td = out_x + (size_t)kBN * kC;      // (B, M, C)

  // ---- workspace carve-out (~0.9 GB) ----
  char* p = (char*)d_ws;
  float*  xn_f   = (float*)bump(p, (size_t)kBN * kC * 4);
  half_t* xn_h   = (half_t*)bump(p, (size_t)kBN * kC * 2);
  half_t* xn2_h  = (half_t*)bump(p, (size_t)kBN * kC * 2);
  half_t* xn3_h  = (half_t*)bump(p, (size_t)kBN * kC * 2);
  float*  qkv    = (float*)bump(p, (size_t)kBN * 3 * kC * 4);
  half_t* wqkv_h = (half_t*)bump(p, (size_t)kC * 3 * kC * 2);
  half_t* wv_h   = (half_t*)bump(p, (size_t)kC * kC * 2);
  half_t* paca_h = (half_t*)bump(p, (size_t)kC * kC * 2);
  half_t* pwin_h = (half_t*)bump(p, (size_t)kC * kC * 2);
  half_t* fc1_h  = (half_t*)bump(p, (size_t)kC * kHID * 2);
  half_t* fc2_h  = (half_t*)bump(p, (size_t)kHID * kC * 2);
  half_t* td_h   = (half_t*)bump(p, (size_t)kB * kM * kC * 2);
  float*  qn     = (float*)bump(p, (size_t)kBN * kRC * 4);
  float*  kn     = (float*)bump(p, (size_t)kB * kM * kRC * 4);
  float*  v_f    = (float*)bump(p, (size_t)kB * kM * kC * 4);
  half_t* v_h    = (half_t*)bump(p, (size_t)kB * kM * kC * 2);
  float*  sim    = (float*)bump(p, (size_t)kBN * kM * 4);
  half_t* sim_h  = (half_t*)bump(p, (size_t)kBN * kM * 2);
  int*    tk_id  = (int*)bump(p, (size_t)kBN * 4);
  int*    sortix = (int*)bump(p, (size_t)kBN * 4);
  int*    bins   = (int*)bump(p, (size_t)kB * kM * 4);
  int*    basep  = (int*)bump(p, (size_t)kB * kM * 4);
  float*  x_atd  = (float*)bump(p, (size_t)kBN * kC * 4);
  half_t* aca_h  = (half_t*)bump(p, (size_t)kBN * kC * 2);
  float*  x_aca  = (float*)bump(p, (size_t)kBN * kC * 4);
  half_t* ow_h   = (half_t*)bump(p, (size_t)kBN * kC * 2);
  float*  owp    = (float*)bump(p, (size_t)kBN * kC * 4);
  float*  y2     = (float*)bump(p, (size_t)kBN * kHID * 4);
  float*  y3     = (float*)bump(p, (size_t)kBN * kHID * 4);
  half_t* y3_h   = (half_t*)bump(p, (size_t)kBN * kHID * 2);
  half_t* ms_h   = (half_t*)bump(p, (size_t)kB * kM * kN * 2);
  float*  tdac   = (float*)bump(p, (size_t)kB * kM * kC * 4);

  // ---- weight / input f32 -> f16 ----
  f2h_kernel<<<g1(kC * 3 * kC, 256), 256, 0, stream>>>(wqkv_w, wqkv_h, kC * 3 * kC);
  f2h_kernel<<<g1(kC * kC, 256), 256, 0, stream>>>(wv_w, wv_h, kC * kC);
  f2h_kernel<<<g1(kC * kC, 256), 256, 0, stream>>>(paca_w, paca_h, kC * kC);
  f2h_kernel<<<g1(kC * kC, 256), 256, 0, stream>>>(pwin_w, pwin_h, kC * kC);
  f2h_kernel<<<g1(kC * kHID, 256), 256, 0, stream>>>(fc1_w, fc1_h, kC * kHID);
  f2h_kernel<<<g1(kHID * kC, 256), 256, 0, stream>>>(fc2_w, fc2_h, kHID * kC);
  f2h_kernel<<<g1(kB * kM * kC, 256), 256, 0, stream>>>(td, td_h, kB * kM * kC);

  // ---- LN1, qkv projection ----
  ln_kernel<<<kBN, 256, 0, stream>>>(x, n1g, n1b, xn_f, xn_h);
  gemm_f16_kernel<<<dim3(3 * kC / 64, kBN / 64), 128, 0, stream>>>(
      xn_h, wqkv_h, wqkv_b, qkv, kBN, 3 * kC, kC, 0, 0);

  // ---- ATD cross attention ----
  proj_rc_norm_kernel<<<g1(kBN, 256), 256, 0, stream>>>(xn_f, wq_w, wq_b, qn, kBN);
  proj_rc_norm_kernel<<<1, 256, 0, stream>>>(td, wk_w, wk_b, kn, kB * kM);
  gemm_f16_kernel<<<dim3(kC / 64, (kB * kM) / 64), 128, 0, stream>>>(
      td_h, wv_h, wv_b, v_f, kB * kM, kC, kC, 0, 0);
  f2h_kernel<<<g1(kB * kM * kC, 256), 256, 0, stream>>>(v_f, v_h, kB * kM * kC);
  atd_attn_kernel<<<kBN / 256, 256, 0, stream>>>(qn, kn, atd_scale, sim, sim_h, tk_id);
  for (int b = 0; b < kB; ++b)
    gemm_f16_kernel<<<dim3(kC / 64, kN / 64), 128, 0, stream>>>(
        sim_h + (size_t)b * kN * kM, v_h + (size_t)b * kM * kC, nullptr,
        x_atd + (size_t)b * kN * kC, kN, kC, kM, 0, 0);

  // ---- stable counting sort by tk_id (64 keys) ----
  fill0_kernel<<<1, 128, 0, stream>>>(bins, kB * kM);
  hist_kernel<<<g1(kBN, 256), 256, 0, stream>>>(tk_id, bins);
  scan_kernel<<<1, 32, 0, stream>>>(bins, basep);
  stable_scatter_kernel<<<kB * kM, 256, 0, stream>>>(tk_id, basep, sortix);

  // ---- AC-MSA grouped attention + projection ----
  attn256_kernel<0><<<kB * kNWIN * kNH, 128, 0, stream>>>(
      qkv, sortix, rpi_sa, amask, rpb_table, logit_scale, aca_h);
  gemm_f16_kernel<<<dim3(kC / 64, kBN / 64), 128, 0, stream>>>(
      aca_h, paca_h, paca_b, x_aca, kBN, kC, kC, 0, 0);

  // ---- shifted-window MSA + projection ----
  attn256_kernel<1><<<kB * kNWIN * kNH, 128, 0, stream>>>(
      qkv, sortix, rpi_sa, amask, rpb_table, logit_scale, ow_h);
  gemm_f16_kernel<<<dim3(kC / 64, kBN / 64), 128, 0, stream>>>(
      ow_h, pwin_h, pwin_b, owp, kBN, kC, kC, 0, 0);

  // ---- residual sum into d_out (x) ----
  residual_sum_kernel<<<g1((size_t)kBN * kC, 256), 256, 0, stream>>>(
      x, owp, x_atd, x_aca, out_x);

  // ---- ConvFFN ----
  ln_kernel<<<kBN, 256, 0, stream>>>(out_x, n2g, n2b, nullptr, xn2_h);
  gemm_f16_kernel<<<dim3(kHID / 64, kBN / 64), 128, 0, stream>>>(
      xn2_h, fc1_h, fc1_b, y2, kBN, kHID, kC, /*gelu*/1, 0);
  dwconv_kernel<<<g1((size_t)kBN * kHID, 256), 256, 0, stream>>>(
      y2, dw_w, dw_b, y3, y3_h);
  gemm_f16_kernel<<<dim3(kC / 64, kBN / 64), 128, 0, stream>>>(
      y3_h, fc2_h, fc2_b, out_x, kBN, kC, kHID, 0, /*accum*/1);

  // ---- token dictionary refinement ----
  ln_kernel<<<kBN, 256, 0, stream>>>(out_x, n3g, n3b, nullptr, xn3_h);
  col_softmax_kernel<<<kB * kM, 256, 0, stream>>>(sim, ms_h);
  for (int b = 0; b < kB; ++b)
    gemm_f16_kernel<<<dim3(kC / 64, 1), 128, 0, stream>>>(
        ms_h + (size_t)b * kM * kN, xn3_h + (size_t)b * kN * kC, nullptr,
        tdac + (size_t)b * kM * kC, kM, kC, kN, 0, 0);
  td_new_kernel<<<g1(kB * kM * kC, 256), 256, 0, stream>>>(td, tdac, sigma, out_td);
}
